// GraphWrapper_77292231459504
// MI455X (gfx1250) — matrix operand order repaired and checked by reading, not compiled, not run on hardware
//
#include <hip/hip_runtime.h>

#define B_  2
#define N_  800
#define F_  7
#define DH_ 32
#define BN_ (B_*N_)

typedef __attribute__((ext_vector_type(16))) _Float16 v16h;
typedef __attribute__((ext_vector_type(8)))  float    v8f;

// gelu(tanh approx) = x * sigmoid(1.59577*(x + 0.044715 x^3))
// branch-free: v_exp_f32 (+ log2e premul inside __expf) + v_rcp_f32
__device__ __forceinline__ float gelu_f(float x){
  const float k2c = 1.5957691216057308f;   // 2*sqrt(2/pi)
  float u = x + 0.044715f*x*x*x;
  float e = __expf(-k2c*u);
  return x * __builtin_amdgcn_rcpf(1.0f + e);
}

// ---------------- embed: h = gelu(x[...,3:] @ W_emb + b_emb); pos = x[...,:3]
__global__ void k_embed(const float* __restrict__ x, const float* __restrict__ Wemb,
                        const float* __restrict__ bemb, float* __restrict__ h,
                        float* __restrict__ pos){
  int idx = blockIdx.x*blockDim.x + threadIdx.x;
  if (idx >= BN_*DH_) return;
  int n = idx & (DH_-1);
  int row = idx >> 5;
  float acc = bemb[n];
  #pragma unroll
  for (int k=0;k<4;++k) acc += x[row*F_ + 3 + k]*Wemb[k*DH_ + n];
  h[idx] = gelu_f(acc);
  if (n < 3) pos[row*3+n] = x[row*F_ + n];
}

// ---------------- mi = h@Wi, mj = h@Wj (rows 0..31 / 32..63 of We1[t])
__global__ void k_mij(const float* __restrict__ h, const float* __restrict__ We1,
                      int t, float* __restrict__ mi, float* __restrict__ mj){
  int idx = blockIdx.x*blockDim.x + threadIdx.x;
  if (idx >= BN_*DH_) return;
  int n = idx & (DH_-1);
  int row = idx >> 5;
  const float* Wi = We1 + (size_t)t*96*DH_;
  const float* Wj = Wi + 32*DH_;
  float ai=0.f, aj=0.f;
  #pragma unroll 8
  for (int k=0;k<DH_;++k){
    float hv = h[(size_t)row*DH_+k];
    ai += hv*Wi[k*DH_+n];
    aj += hv*Wj[k*DH_+n];
  }
  mi[idx]=ai; mj[idx]=aj;
}

__global__ void k_zero(float* __restrict__ agg, float* __restrict__ pd){
  int idx = blockIdx.x*blockDim.x + threadIdx.x;
  if (idx < BN_*DH_) agg[idx] = 0.f;
  else if (idx < BN_*DH_ + BN_*3) pd[idx - BN_*DH_] = 0.f;
}

// ---------------- fused edge MLP: one wave per (b, 16-i tile, 32-j chunk)
__global__ __launch_bounds__(32) void k_edge(
    const float* __restrict__ pos,
    const float* __restrict__ mi, const float* __restrict__ mj,
    const float* __restrict__ We1, const float* __restrict__ be1,
    const float* __restrict__ We2, const float* __restrict__ be2,
    const float* __restrict__ We3, const float* __restrict__ be3,
    const float* __restrict__ Wp1, const float* __restrict__ bp1,
    const float* __restrict__ Wp2, const float* __restrict__ bp2,
    float* __restrict__ agg, float* __restrict__ pd, int t)
{
  const int lane = threadIdx.x;
  const int hi = lane >> 4;     // half-wave id
  const int ln = lane & 15;
  const int b  = blockIdx.z;
  const int it = blockIdx.y;
  const int jc = blockIdx.x;

  __shared__ float sm[16][DH_+1];  // transpose staging C/D-layout -> A-layout

  const float* Wf = We1 + ((size_t)t*96 + 2*DH_)*DH_;
  const float* W2 = We2 + (size_t)t*DH_*DH_;
  const float* W3 = We3 + (size_t)t*DH_*DH_;
  const float* Wq = Wp1 + (size_t)t*DH_*DH_;

  // B-layout weights (wave32 f16 B 32x16: lane n=ln(+16*hf matrix half), K = e + (e<8?0:8) + 8*hi)
  v16h wf[2], w2[2], w3[2], wq[2];
  #pragma unroll
  for (int hf=0; hf<2; ++hf){
    const int n = ln + 16*hf;
    #pragma unroll
    for (int e=0;e<16;++e){
      const int k = e + ((e<8)?0:8) + 8*hi;
      wf[hf][e] = (_Float16)Wf[k*DH_+n];
      w2[hf][e] = (_Float16)W2[k*DH_+n];
      w3[hf][e] = (_Float16)W3[k*DH_+n];
      wq[hf][e] = (_Float16)Wq[k*DH_+n];
    }
  }
  float be1v[2], be2v[2], be3v[2], bp1v[2], wp2v[2];
  #pragma unroll
  for (int hf=0; hf<2; ++hf){
    int n = ln+16*hf;
    be1v[hf]=be1[t*DH_+n]; be2v[hf]=be2[t*DH_+n]; be3v[hf]=be3[t*DH_+n];
    bp1v[hf]=bp1[t*DH_+n]; wp2v[hf]=Wp2[t*DH_+n];
  }
  const float bp2v = bp2[t];

  // per C/D-layout row (M = r + 8*hi): mi values and positions
  float miv[2][8];
  float pr[8][3];
  float pdacc[8][3];
  #pragma unroll
  for (int r=0;r<8;++r){
    const int irow = it*16 + r + 8*hi;
    #pragma unroll
    for (int hf=0; hf<2; ++hf)
      miv[hf][r] = mi[((size_t)(b*N_+irow))*DH_ + ln + 16*hf];
    pr[r][0]=pos[(b*N_+irow)*3+0];
    pr[r][1]=pos[(b*N_+irow)*3+1];
    pr[r][2]=pos[(b*N_+irow)*3+2];
    pdacc[r][0]=0.f; pdacc[r][1]=0.f; pdacc[r][2]=0.f;
  }
  // A-layout row for this lane (M = ln)
  const int iA = it*16 + ln;
  const float pix = pos[(b*N_+iA)*3+0];
  const float piy = pos[(b*N_+iA)*3+1];
  const float piz = pos[(b*N_+iA)*3+2];

  // A-layout K pattern (16-bit A 16x32): lane half hi -> freqs (e&7)+8*hi; e<8 sin, e>=8 cos
  float frq[8];
  #pragma unroll
  for (int m=0;m<8;++m)
    frq[m] = 3.14159265358979323846f * exp2f((float)(m + 8*hi));

  v8f agg0 = {}, agg1 = {};

  for (int jj=0; jj<32; ++jj){
    const int j = jc*32 + jj;
    const float pjx = pos[(b*N_+j)*3+0];
    const float pjy = pos[(b*N_+j)*3+1];
    const float pjz = pos[(b*N_+j)*3+2];
    const float dx = pix-pjx, dy = piy-pjy, dz = piz-pjz;
    const float d2 = dx*dx+dy*dy+dz*dz;

    // Fourier features directly in A layout
    v16h a;
    #pragma unroll
    for (int e=0;e<8;++e) a[e]   = (_Float16)__sinf(d2*frq[e]);
    #pragma unroll
    for (int e=0;e<8;++e) a[e+8] = (_Float16)__cosf(d2*frq[e]);

    const float mj0 = mj[((size_t)(b*N_+j))*DH_ + ln];
    const float mj1 = mj[((size_t)(b*N_+j))*DH_ + ln + 16];
    v8f c0, c1;
    #pragma unroll
    for (int r=0;r<8;++r){ c0[r]=miv[0][r]+mj0+be1v[0]; c1[r]=miv[1][r]+mj1+be1v[1]; }
    v8f d0 = __builtin_amdgcn_wmma_f32_16x16x32_f16(false,a,false,wf[0],(short)0,c0,false,false);
    v8f d1 = __builtin_amdgcn_wmma_f32_16x16x32_f16(false,a,false,wf[1],(short)0,c1,false,false);

    // gelu -> LDS -> A layout
    #pragma unroll
    for (int r=0;r<8;++r){
      sm[r+8*hi][ln]    = gelu_f(d0[r]);
      sm[r+8*hi][ln+16] = gelu_f(d1[r]);
    }
    __syncthreads();
    v16h a1;
    #pragma unroll
    for (int e=0;e<16;++e){
      const int k = e + ((e<8)?0:8) + 8*hi;
      a1[e] = (_Float16)sm[ln][k];
    }
    __syncthreads();

    #pragma unroll
    for (int r=0;r<8;++r){ c0[r]=be2v[0]; c1[r]=be2v[1]; }
    d0 = __builtin_amdgcn_wmma_f32_16x16x32_f16(false,a1,false,w2[0],(short)0,c0,false,false);
    d1 = __builtin_amdgcn_wmma_f32_16x16x32_f16(false,a1,false,w2[1],(short)0,c1,false,false);
    #pragma unroll
    for (int r=0;r<8;++r){
      sm[r+8*hi][ln]    = gelu_f(d0[r]);
      sm[r+8*hi][ln+16] = gelu_f(d1[r]);
    }
    __syncthreads();
    #pragma unroll
    for (int e=0;e<16;++e){
      const int k = e + ((e<8)?0:8) + 8*hi;
      a1[e] = (_Float16)sm[ln][k];
    }
    __syncthreads();

    #pragma unroll
    for (int r=0;r<8;++r){ c0[r]=be3v[0]; c1[r]=be3v[1]; }
    d0 = __builtin_amdgcn_wmma_f32_16x16x32_f16(false,a1,false,w3[0],(short)0,c0,false,false);
    d1 = __builtin_amdgcn_wmma_f32_16x16x32_f16(false,a1,false,w3[1],(short)0,c1,false,false);

    // gelu + diagonal mask + accumulate agg + stage for Wp1
    #pragma unroll
    for (int r=0;r<8;++r){
      const int irow = it*16 + r + 8*hi;
      const float msk = (irow == j) ? 0.f : 1.f;
      const float m0 = msk*gelu_f(d0[r]);
      const float m1 = msk*gelu_f(d1[r]);
      agg0[r] += m0; agg1[r] += m1;
      sm[r+8*hi][ln]    = m0;
      sm[r+8*hi][ln+16] = m1;
    }
    __syncthreads();
    #pragma unroll
    for (int e=0;e<16;++e){
      const int k = e + ((e<8)?0:8) + 8*hi;
      a1[e] = (_Float16)sm[ln][k];
    }
    __syncthreads();

    #pragma unroll
    for (int r=0;r<8;++r){ c0[r]=bp1v[0]; c1[r]=bp1v[1]; }
    d0 = __builtin_amdgcn_wmma_f32_16x16x32_f16(false,a1,false,wq[0],(short)0,c0,false,false);
    d1 = __builtin_amdgcn_wmma_f32_16x16x32_f16(false,a1,false,wq[1],(short)0,c1,false,false);

    // w_p = gelu(.)@Wp2 + bp2 : butterfly reduce over the 16 lanes of each half
    #pragma unroll
    for (int r=0;r<8;++r){
      float s = gelu_f(d0[r])*wp2v[0] + gelu_f(d1[r])*wp2v[1];
      s += __shfl_xor(s, 1, 32);
      s += __shfl_xor(s, 2, 32);
      s += __shfl_xor(s, 4, 32);
      s += __shfl_xor(s, 8, 32);
      const float wp = s + bp2v;
      pdacc[r][0] += (pr[r][0]-pjx)*wp;
      pdacc[r][1] += (pr[r][1]-pjy)*wp;
      pdacc[r][2] += (pr[r][2]-pjz)*wp;
    }
  }

  // reductions across j-chunks
  #pragma unroll
  for (int r=0;r<8;++r){
    const int irow = it*16 + r + 8*hi;
    atomicAdd(&agg[((size_t)(b*N_+irow))*DH_ + ln],      agg0[r]);
    atomicAdd(&agg[((size_t)(b*N_+irow))*DH_ + ln + 16], agg1[r]);
  }
  if (ln == 0){
    #pragma unroll
    for (int r=0;r<8;++r){
      const int irow = it*16 + r + 8*hi;
      atomicAdd(&pd[(b*N_+irow)*3+0], pdacc[r][0]);
      atomicAdd(&pd[(b*N_+irow)*3+1], pdacc[r][1]);
      atomicAdd(&pd[(b*N_+irow)*3+2], pdacc[r][2]);
    }
  }
}

// ---------------- node update + pos update (one wave per node)
__global__ __launch_bounds__(32) void k_node(
    const float* __restrict__ h, const float* __restrict__ agg,
    const float* __restrict__ posin, const float* __restrict__ pd,
    const float* __restrict__ Wn1, const float* __restrict__ bn1,
    const float* __restrict__ Wn2, const float* __restrict__ bn2,
    const float* __restrict__ Wn3, const float* __restrict__ bn3,
    int t, float* __restrict__ hout, float* __restrict__ posout)
{
  const int row = blockIdx.x;
  const int n = threadIdx.x;
  __shared__ float nin[2*DH_];
  __shared__ float u1[DH_];
  __shared__ float u2[DH_];
  nin[n]     = h[(size_t)row*DH_+n];
  nin[n+DH_] = agg[(size_t)row*DH_+n];
  __syncthreads();
  const float* W1 = Wn1 + (size_t)t*2*DH_*DH_;
  float a1 = bn1[t*DH_+n];
  #pragma unroll 8
  for (int k=0;k<2*DH_;++k) a1 += nin[k]*W1[k*DH_+n];
  u1[n] = gelu_f(a1);
  __syncthreads();
  const float* W2p = Wn2 + (size_t)t*DH_*DH_;
  float a2 = bn2[t*DH_+n];
  #pragma unroll 8
  for (int k=0;k<DH_;++k) a2 += u1[k]*W2p[k*DH_+n];
  u2[n] = gelu_f(a2);
  __syncthreads();
  const float* W3p = Wn3 + (size_t)t*DH_*DH_;
  float a3 = bn3[t*DH_+n];
  #pragma unroll 8
  for (int k=0;k<DH_;++k) a3 += u2[k]*W3p[k*DH_+n];
  hout[(size_t)row*DH_+n] = a3;
  if (n<3) posout[row*3+n] = posin[row*3+n] + pd[row*3+n]*(1.0f/(float)(N_-1));
}

// ---------------- pack output: (B,N,3+32)
__global__ void k_pack(const float* __restrict__ pos, const float* __restrict__ h,
                       float* __restrict__ out){
  int idx = blockIdx.x*blockDim.x + threadIdx.x;
  if (idx >= BN_*35) return;
  int c = idx % 35, row = idx / 35;
  out[idx] = (c < 3) ? pos[row*3+c] : h[(size_t)row*DH_ + (c-3)];
}

extern "C" void kernel_launch(void* const* d_in, const int* in_sizes, int n_in,
                              void* d_out, int out_size, void* d_ws, size_t ws_size,
                              hipStream_t stream){
  (void)in_sizes; (void)n_in; (void)out_size; (void)ws_size;
  const float* x    = (const float*)d_in[0];
  const float* Wemb = (const float*)d_in[1];
  const float* bemb = (const float*)d_in[2];
  const float* We1  = (const float*)d_in[3];
  const float* be1  = (const float*)d_in[4];
  const float* We2  = (const float*)d_in[5];
  const float* be2  = (const float*)d_in[6];
  const float* We3  = (const float*)d_in[7];
  const float* be3  = (const float*)d_in[8];
  const float* Wn1  = (const float*)d_in[9];
  const float* bn1  = (const float*)d_in[10];
  const float* Wn2  = (const float*)d_in[11];
  const float* bn2  = (const float*)d_in[12];
  const float* Wn3  = (const float*)d_in[13];
  const float* bn3  = (const float*)d_in[14];
  const float* Wp1  = (const float*)d_in[15];
  const float* bp1  = (const float*)d_in[16];
  const float* Wp2  = (const float*)d_in[17];
  const float* bp2  = (const float*)d_in[18];
  float* out = (float*)d_out;

  float* ws = (float*)d_ws;
  float* hb0 = ws;
  float* hb1 = hb0 + BN_*DH_;
  float* pb0 = hb1 + BN_*DH_;
  float* pb1 = pb0 + BN_*3;
  float* mi  = pb1 + BN_*3;
  float* mj  = mi  + BN_*DH_;
  float* agg = mj  + BN_*DH_;
  float* pd  = agg + BN_*DH_;
  float* hbuf[2] = { hb0, hb1 };
  float* pbuf[2] = { pb0, pb1 };

  k_embed<<<(BN_*DH_+255)/256, 256, 0, stream>>>(x, Wemb, bemb, hbuf[0], pbuf[0]);

  int cur = 0;
  for (int t=0; t<3; ++t){
    int nxt = cur ^ 1;
    k_mij <<<(BN_*DH_+255)/256, 256, 0, stream>>>(hbuf[cur], We1, t, mi, mj);
    k_zero<<<(BN_*DH_+BN_*3+255)/256, 256, 0, stream>>>(agg, pd);
    k_edge<<<dim3(N_/32, N_/16, B_), 32, 0, stream>>>(
        pbuf[cur], mi, mj, We1, be1, We2, be2, We3, be3,
        Wp1, bp1, Wp2, bp2, agg, pd, t);
    k_node<<<BN_, 32, 0, stream>>>(hbuf[cur], agg, pbuf[cur], pd,
        Wn1, bn1, Wn2, bn2, Wn3, bn3, t, hbuf[nxt], pbuf[nxt]);
    cur = nxt;
  }

  k_pack<<<(BN_*35+255)/256, 256, 0, stream>>>(pbuf[cur], hbuf[cur], out);
}
